// BitMultiheadAttention_3453153706561
// MI455X (gfx1250) — compile-verified, matrix operand order checked by reading, not run on hardware
//
#include <hip/hip_runtime.h>
#include <hip/hip_bf16.h>
#include <math.h>


#define B_  2
#define S_  2048
#define D_  2048
#define H_  16
#define HD_ 128
#define M_  (B_ * S_)
#define WN_ (D_ * D_)

typedef __attribute__((ext_vector_type(16))) _Float16 v16h;
typedef __attribute__((ext_vector_type(8)))  _Float16 v8h;
typedef __attribute__((ext_vector_type(8)))  float    v8f;
typedef __attribute__((ext_vector_type(8)))  int      v8i;

static __device__ __forceinline__ v8f zero8f() {
  v8f z;
#pragma unroll
  for (int i = 0; i < 8; ++i) z[i] = 0.0f;
  return z;
}
static __device__ __forceinline__ v8i zero8i() {
  v8i z;
#pragma unroll
  for (int i = 0; i < 8; ++i) z[i] = 0;
  return z;
}

static __device__ __forceinline__ void async_g2l_b128(unsigned lds_off, const void* g) {
  asm volatile("global_load_async_to_lds_b128 %0, %1, off"
               :: "v"(lds_off), "v"((unsigned long long)(uintptr_t)g)
               : "memory");
}
static __device__ __forceinline__ void wait_async0() {
  asm volatile("s_wait_asynccnt 0x0" ::: "memory");
}
static __device__ __forceinline__ unsigned lds_off32(const void* p) {
  return (unsigned)(uintptr_t)p;
}

static __device__ __forceinline__ v8i load_frag_i8(const signed char* base, int stride) {
  const int lane = threadIdx.x & 31;
  const int r    = lane & 15;
  const int koff = (lane >> 4) << 3;
  const signed char* p = base + (size_t)r * stride + koff;
  int2 c0 = *(const int2*)(p +  0);
  int2 c1 = *(const int2*)(p + 16);
  int2 c2 = *(const int2*)(p + 32);
  int2 c3 = *(const int2*)(p + 48);
  v8i f;
  f[0] = c0.x; f[1] = c0.y; f[2] = c1.x; f[3] = c1.y;
  f[4] = c2.x; f[5] = c2.y; f[6] = c3.x; f[7] = c3.y;
  return f;
}

static __device__ __forceinline__ v16h load_frag_f16g(const _Float16* base, int stride) {
  const int lane = threadIdx.x & 31;
  const int r    = lane & 15;
  const int koff = (lane >> 4) << 3;
  const _Float16* p = base + (size_t)r * stride + koff;
  v8h lo = *(const v8h*)(p);
  v8h hi = *(const v8h*)(p + 16);
  v16h f;
#pragma unroll
  for (int i = 0; i < 8; ++i) { f[i] = lo[i]; f[8 + i] = hi[i]; }
  return f;
}

__global__ __launch_bounds__(256)
void subln_quant_kernel(const float* __restrict__ x, signed char* __restrict__ xq,
                        float* __restrict__ rowscale) {
  const int row = blockIdx.x;
  const int tid = threadIdx.x;
  const float* p = x + (size_t)row * D_;
  __shared__ float red[256];

  float sum = 0.f, sumsq = 0.f;
  for (int i = tid; i < D_; i += 256) { float v = p[i]; sum += v; sumsq += v * v; }

  red[tid] = sum; __syncthreads();
  for (int st = 128; st > 0; st >>= 1) { if (tid < st) red[tid] += red[tid + st]; __syncthreads(); }
  const float mean = red[0] / (float)D_;
  __syncthreads();

  red[tid] = sumsq; __syncthreads();
  for (int st = 128; st > 0; st >>= 1) { if (tid < st) red[tid] += red[tid + st]; __syncthreads(); }
  const float var  = red[0] / (float)D_ - mean * mean;
  __syncthreads();
  const float rstd = rsqrtf(var + 1e-5f);

  float amax = 0.f;
  for (int i = tid; i < D_; i += 256) amax = fmaxf(amax, fabsf((p[i] - mean) * rstd));
  red[tid] = amax; __syncthreads();
  for (int st = 128; st > 0; st >>= 1) { if (tid < st) red[tid] = fmaxf(red[tid], red[tid + st]); __syncthreads(); }
  amax = fmaxf(red[0], 1e-5f);

  const float qs = 127.f / amax;
  for (int i = tid; i < D_; i += 256) {
    float q = rintf((p[i] - mean) * rstd * qs);
    q = fminf(fmaxf(q, -128.f), 127.f);
    xq[(size_t)row * D_ + i] = (signed char)q;
  }
  if (tid == 0) rowscale[row] = amax / 127.f;
}

__global__ __launch_bounds__(256)
void absmean_partial_kernel(const float* __restrict__ w, float* __restrict__ partial) {
  __shared__ float red[256];
  const int tid = threadIdx.x;
  float s = 0.f;
  for (int i = blockIdx.x * 256 + tid; i < WN_; i += gridDim.x * 256) s += fabsf(w[i]);
  red[tid] = s; __syncthreads();
  for (int st = 128; st > 0; st >>= 1) { if (tid < st) red[tid] += red[tid + st]; __syncthreads(); }
  if (tid == 0) partial[blockIdx.x] = red[0];
}

__global__ __launch_bounds__(256)
void absmean_final_kernel(const float* __restrict__ partial, float* __restrict__ ams) {
  __shared__ float red[256];
  const int tid  = threadIdx.x;
  const int widx = blockIdx.x;
  float s = 0.f;
  for (int i = tid; i < 1024; i += 256) s += partial[widx * 1024 + i];
  red[tid] = s; __syncthreads();
  for (int st = 128; st > 0; st >>= 1) { if (tid < st) red[tid] += red[tid + st]; __syncthreads(); }
  if (tid == 0) ams[widx] = red[0] / (float)WN_;
}

__global__ __launch_bounds__(256)
void weight_quant_kernel(const float* __restrict__ w, const float* __restrict__ ams, int widx,
                         signed char* __restrict__ w8) {
  const float inv = 1.f / (ams[widx] + 1e-5f);
  for (int i = blockIdx.x * 256 + threadIdx.x; i < WN_; i += gridDim.x * 256) {
    float q = rintf(w[i] * inv);
    q = fminf(fmaxf(q, -1.f), 1.f);
    w8[i] = (signed char)q;
  }
}

__global__ __launch_bounds__(256)
void bitlin_gemm_kernel(const signed char* __restrict__ xq8,
                        const float* __restrict__ rowscale,
                        const signed char* __restrict__ w8,
                        const float* __restrict__ ams, int am_idx,
                        int mode, float extra,
                        float* __restrict__ out_f32,
                        _Float16* __restrict__ out_h) {
  const int w    = threadIdx.x >> 5;
  const int wr   = w & 3, wc = w >> 2;
  const int row0 = (blockIdx.y * 4 + wr) * 16;
  const int col0 = (blockIdx.x * 2 + wc) * 16;
  const float am = ams[am_idx];

  const signed char* arow = xq8 + (size_t)row0 * D_;
  const signed char* brow = w8  + (size_t)col0 * D_;

  v8i acc = zero8i();
  v8i a = load_frag_i8(arow, D_);
  v8i b = load_frag_i8(brow, D_);
  for (int k0 = 64; k0 < D_; k0 += 64) {
    v8i an = load_frag_i8(arow + k0, D_);
    v8i bn = load_frag_i8(brow + k0, D_);
    __builtin_prefetch(arow + k0 + 64, 0, 1);
    acc = __builtin_amdgcn_wmma_i32_16x16x64_iu8(true, a, true, b, acc, false, false);
    a = an; b = bn;
  }
  acc = __builtin_amdgcn_wmma_i32_16x16x64_iu8(true, a, true, b, acc, false, false);

  const int lane = threadIdx.x & 31;
  const int col  = lane & 15;
  const int rb   = (lane >> 4) << 3;
#pragma unroll
  for (int j = 0; j < 8; ++j) {
    const int r = row0 + rb + j;
    const int c = col0 + col;
    const float v = (float)acc[j] * rowscale[r] * am * extra;
    if (mode == 0) {
      out_f32[(size_t)r * D_ + c] = v;
    } else {
      const int b_ = r >> 11, s_ = r & (S_ - 1);
      const int h_ = c >> 7,  d_ = c & (HD_ - 1);
      if (mode == 1)
        out_h[(((size_t)b_ * H_ + h_) * S_ + s_) * HD_ + d_] = (_Float16)v;
      else
        out_h[(((size_t)b_ * H_ + h_) * HD_ + d_) * S_ + s_] = (_Float16)v;
    }
  }
}

__global__ __launch_bounds__(128)
void attn_kernel(const _Float16* __restrict__ qf,
                 const _Float16* __restrict__ kf,
                 const _Float16* __restrict__ vT,
                 float* __restrict__ out) {
  const int bh   = blockIdx.y;
  const int b    = bh >> 4, h = bh & 15;
  const int wave = threadIdx.x >> 5;
  const int lane = threadIdx.x & 31;
  const int q0b  = blockIdx.x * 64;
  const int q0   = q0b + wave * 16;

  const _Float16* qbase = qf + ((size_t)bh * S_ + q0) * HD_;
  const _Float16* kbase = kf + (size_t)bh * S_ * HD_;
  const _Float16* vbase = vT + (size_t)bh * HD_ * S_;

  __shared__ __align__(16) _Float16 ktile[32][HD_];
  __shared__ __align__(16) _Float16 vtile[HD_][32];
  __shared__ __align__(16) _Float16 pbuf[4][16][40];

  v16h qa[4];
#pragma unroll
  for (int c = 0; c < 4; ++c) qa[c] = load_frag_f16g(qbase + c * 32, HD_);

  v8f o[8];
#pragma unroll
  for (int t = 0; t < 8; ++t) o[t] = zero8f();

  float mst[8], lst[8];
#pragma unroll
  for (int j = 0; j < 8; ++j) { mst[j] = -__builtin_inff(); lst[j] = 0.f; }

  const int col       = lane & 15;
  const int rb        = (lane >> 4) << 3;
  const int my_ntile  = (q0 + 47) >> 5;
  const int ntile_max = (q0b + 95) >> 5;

  const unsigned klds = lds_off32(&ktile[0][0]);
  const unsigned vlds = lds_off32(&vtile[0][0]);

  for (int jt = 0; jt < ntile_max; ++jt) {
    const int key0 = jt * 32;

    const char* kg = (const char*)(kbase + (size_t)key0 * HD_);
#pragma unroll
    for (int i = 0; i < 4; ++i) {
      const int c = (int)threadIdx.x + i * 128;
      async_g2l_b128(klds + c * 16, kg + c * 16);
    }
#pragma unroll
    for (int i = 0; i < 4; ++i) {
      const int c = (int)threadIdx.x + i * 128;
      const int d = c >> 2, cc = c & 3;
      async_g2l_b128(vlds + d * 64 + cc * 16,
                     (const char*)(vbase + (size_t)d * S_ + key0) + cc * 16);
    }
    wait_async0();
    __syncthreads();

    if (jt < my_ntile) {
      v8f s0 = zero8f(), s1 = zero8f();
#pragma unroll
      for (int c = 0; c < 4; ++c) {
        v16h kb0 = load_frag_f16g(&ktile[0][c * 32],  HD_);
        v16h kb1 = load_frag_f16g(&ktile[16][c * 32], HD_);
        s0 = __builtin_amdgcn_wmma_f32_16x16x32_f16(false, qa[c], false, kb0, (short)0, s0, false, false);
        s1 = __builtin_amdgcn_wmma_f32_16x16x32_f16(false, qa[c], false, kb1, (short)0, s1, false, false);
      }
#pragma unroll
      for (int j = 0; j < 8; ++j) {
        const int qrow = q0 + rb + j;
        float a0 = (key0 + col      > qrow) ? -__builtin_inff() : s0[j];
        float a1 = (key0 + 16 + col > qrow) ? -__builtin_inff() : s1[j];
        float tm = fmaxf(a0, a1);
#pragma unroll
        for (int mk = 8; mk >= 1; mk >>= 1) tm = fmaxf(tm, __shfl_xor(tm, mk, 32));
        const float mn   = fmaxf(mst[j], tm);
        const float corr = __expf(mst[j] - mn);
        const float p0   = __expf(a0 - mn);
        const float p1   = __expf(a1 - mn);
        float rs = p0 + p1;
#pragma unroll
        for (int mk = 8; mk >= 1; mk >>= 1) rs += __shfl_xor(rs, mk, 32);
        lst[j] = lst[j] * corr + rs;
        mst[j] = mn;
#pragma unroll
        for (int t = 0; t < 8; ++t) o[t][j] *= corr;
        pbuf[wave][rb + j][col]      = (_Float16)p0;
        pbuf[wave][rb + j][16 + col] = (_Float16)p1;
      }
      v16h pa = load_frag_f16g(&pbuf[wave][0][0], 40);
#pragma unroll
      for (int t = 0; t < 8; ++t) {
        v16h vb = load_frag_f16g(&vtile[t * 16][0], 32);
        o[t] = __builtin_amdgcn_wmma_f32_16x16x32_f16(false, pa, false, vb, (short)0, o[t], false, false);
      }
    }
    __syncthreads();
  }

#pragma unroll
  for (int j = 0; j < 8; ++j) {
    const float inv  = 1.0f / lst[j];
    const int   qrow = q0 + rb + j;
#pragma unroll
    for (int t = 0; t < 8; ++t)
      out[((size_t)b * S_ + qrow) * D_ + h * HD_ + t * 16 + col] = o[t][j] * inv;
  }
}

extern "C" void kernel_launch(void* const* d_in, const int* in_sizes, int n_in,
                              void* d_out, int out_size, void* d_ws, size_t ws_size,
                              hipStream_t stream) {
  (void)in_sizes; (void)n_in; (void)out_size; (void)ws_size;
  const float* x  = (const float*)d_in[0];
  const float* wlist[4] = { (const float*)d_in[1], (const float*)d_in[2],
                            (const float*)d_in[3], (const float*)d_in[4] };

  char* ws = (char*)d_ws;
  size_t off = 0;
  auto alloc = [&](size_t bytes) -> char* {
    char* p = ws + off;
    off = (off + bytes + 255) & ~(size_t)255;
    return p;
  };

  signed char* xq8   = (signed char*)alloc((size_t)M_ * D_);
  float*       rs1   = (float*)alloc((size_t)M_ * 4);
  signed char* w8[4];
  for (int i = 0; i < 4; ++i) w8[i] = (signed char*)alloc((size_t)WN_);
  float*       ams     = (float*)alloc(4 * 4);
  float*       partial = (float*)alloc(4 * 1024 * 4);
  _Float16*    qfb   = (_Float16*)alloc((size_t)M_ * D_ * 2);
  _Float16*    kfb   = (_Float16*)alloc((size_t)M_ * D_ * 2);
  _Float16*    vTb   = (_Float16*)alloc((size_t)M_ * D_ * 2);
  float*       attn  = (float*)alloc((size_t)M_ * D_ * 4);
  signed char* xq8b  = (signed char*)alloc((size_t)M_ * D_);
  float*       rs2   = (float*)alloc((size_t)M_ * 4);

  subln_quant_kernel<<<M_, 256, 0, stream>>>(x, xq8, rs1);

  for (int i = 0; i < 4; ++i)
    absmean_partial_kernel<<<1024, 256, 0, stream>>>(wlist[i], partial + i * 1024);
  absmean_final_kernel<<<4, 256, 0, stream>>>(partial, ams);
  for (int i = 0; i < 4; ++i)
    weight_quant_kernel<<<4096, 256, 0, stream>>>(wlist[i], ams, i, w8[i]);

  dim3 gg(D_ / 32, M_ / 64);
  const float qscale = 0.08838834764831845f;
  bitlin_gemm_kernel<<<gg, 256, 0, stream>>>(xq8, rs1, w8[0], ams, 0, 1, qscale, nullptr, qfb);
  bitlin_gemm_kernel<<<gg, 256, 0, stream>>>(xq8, rs1, w8[1], ams, 1, 1, 1.0f,   nullptr, kfb);
  bitlin_gemm_kernel<<<gg, 256, 0, stream>>>(xq8, rs1, w8[2], ams, 2, 2, 1.0f,   nullptr, vTb);

  attn_kernel<<<dim3(S_ / 64, B_ * H_), 128, 0, stream>>>(qfb, kfb, vTb, attn);

  subln_quant_kernel<<<M_, 256, 0, stream>>>(attn, xq8b, rs2);
  bitlin_gemm_kernel<<<gg, 256, 0, stream>>>(xq8b, rs2, w8[3], ams, 3, 0, 1.0f, (float*)d_out, nullptr);
}
